// SelfAttention2D_3865470566722
// MI455X (gfx1250) — matrix operand order repaired and checked by reading, not compiled, not run on hardware
//
#include <hip/hip_runtime.h>

typedef __attribute__((ext_vector_type(16))) __bf16 bf16x16;
typedef __attribute__((ext_vector_type(8)))  __bf16 bf16x8;
typedef __attribute__((ext_vector_type(8)))  float  f32x8;
typedef __attribute__((ext_vector_type(4)))  unsigned int u32x4;
typedef __attribute__((ext_vector_type(8)))  int i32x8;
typedef __attribute__((ext_vector_type(4)))  int i32x4;

#define WMMA_BF16F32 __builtin_amdgcn_wmma_f32_16x16x32_bf16

namespace {
constexpr int kC = 256;   // channels
constexpr int kD = 32;    // Cqk
constexpr int kN = 4096;  // 64*64 spatial
constexpr int KSTR = 40;  // klds row stride (bf16): 80B, 16B aligned, conflict-free
constexpr int VSTR = 72;  // vlds row stride (bf16): 144B, 16B aligned, conflict-free
}

// Assemble a v16bf A/B operand from two contiguous 8x bf16 (16B) chunks.
__device__ __forceinline__ bf16x16 ld_pair16(const __bf16* p0, const __bf16* p1) {
    bf16x8 a0 = *reinterpret_cast<const bf16x8*>(p0);
    bf16x8 a1 = *reinterpret_cast<const bf16x8*>(p1);
    bf16x16 a;
#pragma unroll
    for (int i = 0; i < 8; ++i) { a[i] = a0[i]; a[8 + i] = a1[i]; }
    return a;
}

// ---------------------------------------------------------------------------
// TDM: 2D tile load global->LDS with row padding (D# groups built in SGPRs).
//   flags_w0 = data_size | pad_enable | pad_interval | pad_amount
// ---------------------------------------------------------------------------
__device__ __forceinline__ void tdm_load_2d(
    unsigned lds_byte, const void* gaddr,
    unsigned tensor_d0, unsigned tensor_d1, unsigned stride0,
    unsigned tile_d0, unsigned tile_d1, unsigned flags_w0)
{
    unsigned long long ga = (unsigned long long)(uintptr_t)gaddr;
    u32x4 g0;
    g0[0] = 1u;                                            // count=1, user mode
    g0[1] = lds_byte;                                      // lds_addr
    g0[2] = (unsigned)(ga & 0xffffffffu);                  // global_addr[31:0]
    g0[3] = (unsigned)((ga >> 32) & 0x01ffffffu)           // global_addr[56:32]
          | (2u << 30);                                    // type=2 (image)
    i32x8 g1;
    g1[0] = (int)flags_w0;                                 // mask=0,data_size,pad
    g1[1] = (int)((tensor_d0 & 0xffffu) << 16);            // tensor_dim0 lo16
    g1[2] = (int)((tensor_d0 >> 16) | ((tensor_d1 & 0xffffu) << 16));
    g1[3] = (int)((tensor_d1 >> 16) | (tile_d0 << 16));    // tile_dim0
    g1[4] = (int)tile_d1;                                  // tile_dim1, tile_dim2=0
    g1[5] = (int)stride0;                                  // tensor_dim0_stride lo32
    g1[6] = 0;                                             // stride0 hi16 | dim1_stride lo16
    g1[7] = 0;
    i32x4 z4 = {};
    i32x8 z8 = {};
    // 6-arg form (this toolchain): (g0, g1, g2, g3, g4, cpol)
    __builtin_amdgcn_tensor_load_to_lds(g0, g1, z4, z4, z8, 0);
}

// dwords-per-row=32 (enc 4), pad 4 dwords (enc 3); data_size=2B; pad_enable
#define VT_FLAGS ((1u << 16) | (1u << 20) | (4u << 22) | (3u << 25))
// dwords-per-row=16 (enc 3), pad 4 dwords (enc 3)
#define K_FLAGS  ((1u << 16) | (1u << 20) | (3u << 22) | (3u << 25))

// ---------------------------------------------------------------------------
// Kernel 0: convert concatenated [Wq;Wk;Wv] (320x256 f32) to bf16 once.
// ---------------------------------------------------------------------------
__global__ __launch_bounds__(256) void wcvt_kernel(
    const float* __restrict__ Wq, const float* __restrict__ Wk,
    const float* __restrict__ Wv, __bf16* __restrict__ Wb)
{
    const int i4 = (blockIdx.x * 256 + threadIdx.x) * 4;   // 80 blocks * 256 * 4 = 81920
    const float* src;
    if (i4 < 32 * 256)      src = Wq + i4;
    else if (i4 < 64 * 256) src = Wk + (i4 - 32 * 256);
    else                    src = Wv + (i4 - 64 * 256);
    float4 v = *reinterpret_cast<const float4*>(src);
    __bf16* dst = Wb + i4;
    dst[0] = (__bf16)v.x; dst[1] = (__bf16)v.y;
    dst[2] = (__bf16)v.z; dst[3] = (__bf16)v.w;
}

// ---------------------------------------------------------------------------
// Kernel 1: fused q/k/v projections (1x1 convs) via bf16 WMMA.
//   Q[b][n][32], K[b][n][32] (bf16) ; Vt[b][c][n] (bf16, transposed for PV).
// Block: 128 threads (4 waves), 64 spatial rows of one batch.
// ---------------------------------------------------------------------------
__global__ __launch_bounds__(128) void proj_kernel(
    const float* __restrict__ x, const __bf16* __restrict__ Wb,
    const float* __restrict__ bq, const float* __restrict__ bk,
    const float* __restrict__ bv,
    __bf16* __restrict__ Qw, __bf16* __restrict__ Kw, __bf16* __restrict__ Vt)
{
    __shared__ float xs[kC * 64];  // x tile: [c][n] (64KB)

    const int b   = blockIdx.x >> 6;
    const int n0  = (blockIdx.x & 63) << 6;
    const int tid = threadIdx.x;

    for (int i = tid; i < kC * 16; i += 128) {
        const int c = i >> 4, j = i & 15;
        float4 v = reinterpret_cast<const float4*>(x + (size_t)(b * kC + c) * kN + n0)[j];
        reinterpret_cast<float4*>(&xs[c * 64])[j] = v;
    }
    __syncthreads();

    const int wave = tid >> 5;
    const int lane = tid & 31;
    const int l16  = lane & 15;
    const int hi   = lane >> 4;
    const int nl   = wave * 16 + l16;

    f32x8 acc[20];
#pragma unroll
    for (int cc = 0; cc < 20; ++cc) acc[cc] = (f32x8){};

    for (int ks = 0; ks < 8; ++ks) {          // K-dim: 8 chunks of 32 channels
        bf16x16 a;
#pragma unroll
        for (int i = 0; i < 16; ++i) {
            const int kl = (hi << 4) + i;
            a[i] = (__bf16)xs[(ks * 32 + kl) * 64 + nl];
        }
        // bf16 weight row base for this lane: col = cc*16+l16, kdim = ks*32+16*hi
        const __bf16* wb0 = Wb + (size_t)l16 * kC + ks * 32 + (hi << 4);
        bf16x16 bb = ld_pair16(wb0, wb0 + 8);
#pragma unroll
        for (int cc = 0; cc < 20; ++cc) {     // pipelined: load cc+1 before wmma(cc)
            bf16x16 nxt = bb;
            if (cc < 19) {
                const __bf16* wp = wb0 + (size_t)(cc + 1) * 16 * kC;
                nxt = ld_pair16(wp, wp + 8);
            }
            acc[cc] = WMMA_BF16F32(false, a, false, bb, (short)0, acc[cc], false, false);
            bb = nxt;
        }
    }

    // Epilogue: bias, convert to bf16, scatter to Q/K/Vt.
#pragma unroll
    for (int cc = 0; cc < 20; ++cc) {
        const int outcol = cc * 16 + l16;
        float bias;
        if (cc < 2)      bias = bq[outcol];
        else if (cc < 4) bias = bk[outcol - 32];
        else             bias = bv[outcol - 64];

        const int nb = n0 + wave * 16 + (hi << 3);
        if (cc < 4) {
            const int d = (cc < 2) ? outcol : (outcol - 32);
            __bf16* dst = (cc < 2) ? Qw : Kw;
#pragma unroll
            for (int r = 0; r < 8; ++r)
                dst[(size_t)(b * kN + nb + r) * kD + d] = (__bf16)(acc[cc][r] + bias);
        } else {
            const int ch = outcol - 64;
            bf16x8 pk;
#pragma unroll
            for (int r = 0; r < 8; ++r) pk[r] = (__bf16)(acc[cc][r] + bias);
            *reinterpret_cast<bf16x8*>(&Vt[(size_t)(b * kC + ch) * kN + nb]) = pk;
        }
    }
}

// ---------------------------------------------------------------------------
// Kernel 2: flash attention, transposed-S formulation, TDM double buffering
// in both passes. Block: 128 threads (4 waves), 64 query rows of one batch.
// ---------------------------------------------------------------------------
__global__ __launch_bounds__(128) void attn_kernel(
    const __bf16* __restrict__ Qw, const __bf16* __restrict__ Kw,
    const __bf16* __restrict__ Vt, const float* __restrict__ gammap,
    float* __restrict__ out)
{
    __shared__ alignas(16) __bf16 klds[2][64 * KSTR];   // 2 x  5 KB
    __shared__ alignas(16) __bf16 vlds[2][kC * VSTR];   // 2 x 36 KB

    const int b   = blockIdx.x >> 6;
    const int n0  = (blockIdx.x & 63) << 6;
    const int tid = threadIdx.x;
    const int wave = tid >> 5, lane = tid & 31, l16 = lane & 15, hi = lane >> 4;

    // Resident Q B-operand: lane's column = query row n0+wave*16+l16.
    const int nq = n0 + wave * 16 + l16;
    const __bf16* qrow = Qw + (size_t)(b * kN + nq) * kD + (hi << 3);
    const bf16x16 qb = ld_pair16(qrow, qrow + 16);

    // ---- pass 1: row max over all 4096 keys (TDM double-buffered K) ----
    float m = -3.0e38f;
    if (wave == 0) {
        tdm_load_2d((unsigned)(uintptr_t)&klds[0][0],
                    Kw + (size_t)b * kN * kD, kD, 64, kD, kD, 64, K_FLAGS);
        __builtin_amdgcn_s_wait_tensorcnt(0);
    }
    __syncthreads();
    for (int kb = 0; kb < kN; kb += 64) {
        const int p = (kb >> 6) & 1;
        if (wave == 0 && (kb + 64) < kN)
            tdm_load_2d((unsigned)(uintptr_t)&klds[1 - p][0],
                        Kw + (size_t)(b * kN + kb + 64) * kD, kD, 64, kD, kD, 64, K_FLAGS);
#pragma unroll
        for (int c16 = 0; c16 < 4; ++c16) {
            const __bf16* kp = &klds[p][(c16 * 16 + l16) * KSTR + (hi << 3)];
            bf16x16 ka = ld_pair16(kp, kp + 16);
            f32x8 z = {};
            f32x8 st = WMMA_BF16F32(false, ka, false, qb, (short)0, z, false, false);
#pragma unroll
            for (int r = 0; r < 8; ++r) m = fmaxf(m, st[r]);
        }
        if (wave == 0) __builtin_amdgcn_s_wait_tensorcnt(0);
        __syncthreads();
    }
    m = fmaxf(m, __shfl_xor(m, 16, 32));

    // ---- pass 2: P = exp(S-m), accumulate P.V and row sums ----
    float lsum = 0.0f;
    f32x8 acc[16];
#pragma unroll
    for (int cc = 0; cc < 16; ++cc) acc[cc] = (f32x8){};

    // TDM prologue: tile 0 into buffer 0 (barrier above guarantees pass-1 done).
    if (wave == 0) {
        tdm_load_2d((unsigned)(uintptr_t)&klds[0][0],
                    Kw + (size_t)b * kN * kD, kD, 64, kD, kD, 64, K_FLAGS);
        tdm_load_2d((unsigned)(uintptr_t)&vlds[0][0],
                    Vt + (size_t)b * kC * kN, kN, kC, kN, 64, kC, VT_FLAGS);
        __builtin_amdgcn_s_wait_tensorcnt(0);
    }
    __syncthreads();

    for (int kb = 0; kb < kN; kb += 64) {
        const int p = (kb >> 6) & 1;
        // Kick off DMA of tile kb+64 into the other buffer while we compute.
        if (wave == 0 && (kb + 64) < kN) {
            tdm_load_2d((unsigned)(uintptr_t)&klds[1 - p][0],
                        Kw + (size_t)(b * kN + kb + 64) * kD, kD, 64, kD, kD, 64, K_FLAGS);
            tdm_load_2d((unsigned)(uintptr_t)&vlds[1 - p][0],
                        Vt + (size_t)b * kC * kN + (kb + 64), kN, kC, kN, 64, kC, VT_FLAGS);
        }
        const __bf16* kl = &klds[p][0];
        const __bf16* vl = &vlds[p][0];

#pragma unroll
        for (int c32 = 0; c32 < 2; ++c32) {
            const __bf16* kp0 = kl + (c32 * 32 +      l16) * KSTR + (hi << 3);
            const __bf16* kp1 = kl + (c32 * 32 + 16 + l16) * KSTR + (hi << 3);
            bf16x16 ka0 = ld_pair16(kp0, kp0 + 16);
            bf16x16 ka1 = ld_pair16(kp1, kp1 + 16);
            f32x8 z = {};
            f32x8 st0 = WMMA_BF16F32(false, ka0, false, qb, (short)0, z, false, false);
            f32x8 st1 = WMMA_BF16F32(false, ka1, false, qb, (short)0, z, false, false);

            // S^T -> P, packed directly into P.V A-operand layout (lane-local).
            bf16x16 ap;
            float ps = 0.0f;
#pragma unroll
            for (int r = 0; r < 8; ++r) {
                const float p0 = __expf(st0[r] - m);
                const float p1 = __expf(st1[r] - m);
                ps += p0 + p1;
                ap[r]     = (__bf16)p0;
                ap[8 + r] = (__bf16)p1;
            }
            lsum += ps;

            const __bf16* vbase = vl + l16 * VSTR + c32 * 32 + (hi << 3);
            bf16x16 vb = ld_pair16(vbase, vbase + 16);
#pragma unroll
            for (int cc = 0; cc < 16; ++cc) {  // pipelined LDS B loads
                bf16x16 nxt = vb;
                if (cc < 15) {
                    const __bf16* vp = vbase + (size_t)(cc + 1) * 16 * VSTR;
                    nxt = ld_pair16(vp, vp + 16);
                }
                acc[cc] = WMMA_BF16F32(false, ap, false, vb, (short)0, acc[cc], false, false);
                vb = nxt;
            }
        }

        if (wave == 0) __builtin_amdgcn_s_wait_tensorcnt(0);
        __syncthreads();   // next tile resident; everyone done with buffer p
    }

    // ---- normalize + store out[b][ch][n] (f32) ----
    const float l = lsum + __shfl_xor(lsum, 16, 32);
    const float gamma = gammap[0];
    float scale[8];
#pragma unroll
    for (int r = 0; r < 8; ++r)
        scale[r] = gamma / __shfl(l, r + (hi << 3), 32);

    const int nb = n0 + wave * 16 + (hi << 3);
#pragma unroll
    for (int cc = 0; cc < 16; ++cc) {
        const int ch = cc * 16 + l16;
        float4 f0 = { acc[cc][0] * scale[0], acc[cc][1] * scale[1],
                      acc[cc][2] * scale[2], acc[cc][3] * scale[3] };
        float4 f1 = { acc[cc][4] * scale[4], acc[cc][5] * scale[5],
                      acc[cc][6] * scale[6], acc[cc][7] * scale[7] };
        float4* op = reinterpret_cast<float4*>(out + (size_t)(b * kC + ch) * kN + nb);
        op[0] = f0; op[1] = f1;
    }
}

// ---------------------------------------------------------------------------
extern "C" void kernel_launch(void* const* d_in, const int* in_sizes, int n_in,
                              void* d_out, int out_size, void* d_ws, size_t ws_size,
                              hipStream_t stream)
{
    const float* x     = (const float*)d_in[0];
    const float* Wq    = (const float*)d_in[1];
    const float* bq    = (const float*)d_in[2];
    const float* Wk    = (const float*)d_in[3];
    const float* bk    = (const float*)d_in[4];
    const float* Wv    = (const float*)d_in[5];
    const float* bv    = (const float*)d_in[6];
    const float* gamma = (const float*)d_in[7];

    __bf16* Qw = (__bf16*)d_ws;                       // 4*4096*32 bf16 = 1 MB
    __bf16* Kw = Qw + (size_t)4 * kN * kD;            // 1 MB
    __bf16* Vt = Kw + (size_t)4 * kN * kD;            // 8 MB
    __bf16* Wb = Vt + (size_t)4 * kC * kN;            // 320*256 bf16 = 160 KB

    wcvt_kernel<<<80, 256, 0, stream>>>(Wq, Wk, Wv, Wb);
    proj_kernel<<<256, 128, 0, stream>>>(x, Wb, bq, bk, bv, Qw, Kw, Vt);
    attn_kernel<<<256, 128, 0, stream>>>(Qw, Kw, Vt, gamma, (float*)d_out);
}
